// EnhancingFeatureModule_26405458936158
// MI455X (gfx1250) — matrix operand order repaired and checked by reading, not compiled, not run on hardware
//
#include <hip/hip_runtime.h>
#include <hip/hip_bf16.h>

typedef __attribute__((ext_vector_type(16))) _Float16 v16h;
typedef __attribute__((ext_vector_type(8)))  float    v8f;

#define NPTS  2048
#define BATCH 2
#define KNN   16
#define NKCOL (NPTS*KNN)   // 32768
#define EDGE_C 1923

// ---------------------------------------------------------------------------
// WMMA fragment loaders (wave32; A and B both use contiguous k = 16*(lane/16) + e)
// A tile in LDS: row-major [m][k]  (stride 40 halves = 80B, 16B aligned)
// B tile in LDS: col-major [n][k]  (stride 40 halves) -> fragment = 16
//   contiguous halves => 2x ds_load_b128 per fragment.
// ---------------------------------------------------------------------------
__device__ __forceinline__ v16h frag_a(const _Float16* As, int mbase, int lane) {
  int m  = mbase + (lane & 15);
  int kb = (lane >> 4) * 16;
  const _Float16* p = As + m*40;
  v16h a;
#pragma unroll
  for (int e = 0; e < 8; ++e) a[e]   = p[kb + e];
#pragma unroll
  for (int e = 0; e < 8; ++e) a[8+e] = p[kb + 8 + e];
  return a;
}
__device__ __forceinline__ v16h frag_b(const _Float16* Bs, int nbase, int lane) {
  int n  = nbase + (lane & 15);
  int kb = (lane >> 4) * 16;
  const _Float16* p = Bs + n*40 + kb;   // 16 contiguous halves, 16B aligned
  v16h v;
#pragma unroll
  for (int e = 0; e < 16; ++e) v[e] = p[e];
  return v;
}

// ---------------------------------------------------------------------------
// Generic WMMA GEMM:  Y[b,m,col] = sum_k W[(b)m,k] * X(b,k,col) + bias[m]
// MODE 0: X dense row-major (ldx row stride, xbs batch stride)
// MODE 1: fused edge-feature loader (center | neighbor | relative xyz);
//         per-block kNN indices cached in LDS before the K loop.
// Block: 128 threads = 4 waves; block tile 64x64; each wave 32x32 (2x2 WMMA).
// Staging: clamped addresses + two-phase (batch 16 loads, then cvt+store) so
// the compiler issues all global loads before a single wait.
// ---------------------------------------------------------------------------
template<int MODE>
__global__ __launch_bounds__(128)
void k_gemm(const float* __restrict__ Wm, long long wbs, const float* __restrict__ bias,
            const float* __restrict__ X, long long xbs, int ldx,
            float* __restrict__ Y, long long ybs, int ldy,
            int M, int K, int NC,
            const float* __restrict__ featT, const int* __restrict__ nnidx,
            const float* __restrict__ xyzT) {
  __shared__ __align__(16) _Float16 As[64*40];   // [m][k]
  __shared__ __align__(16) _Float16 Bs[64*40];   // [n][k]  (fragment-major)
  __shared__ int nnCache[64];
  const int tid  = threadIdx.x;
  const int wave = tid >> 5, lane = tid & 31;
  const int wm = wave >> 1, wn = wave & 1;
  const int b  = blockIdx.z;
  const int mBase = blockIdx.y * 64;
  const int nBase = blockIdx.x * 64;
  const float* Wp = Wm + (long long)b * wbs;

  if (MODE == 1) {
    if (tid < 64) {
      int col = nBase + tid;
      nnCache[tid] = nnidx[((long long)b*NPTS + (col >> 4))*KNN + (col & 15)];
    }
  }

  v8f acc00 = {0.f,0.f,0.f,0.f,0.f,0.f,0.f,0.f};
  v8f acc01 = acc00, acc10 = acc00, acc11 = acc00;

  for (int k0 = 0; k0 < K; k0 += 32) {
    __syncthreads();
    // ---- stage A (weights) 64x32: batch 16 clamped loads, then cvt+store
    {
      float va[16];
#pragma unroll
      for (int it = 0; it < 16; ++it) {
        int i = tid + it*128;
        int r = i >> 5, c = i & 31;
        int m = mBase + r, k = k0 + c;
        bool ok = (m < M) && (k < K);
        long long a = ok ? ((long long)m*K + k) : 0LL;
        va[it] = Wp[a];
      }
#pragma unroll
      for (int it = 0; it < 16; ++it) {
        int i = tid + it*128;
        int r = i >> 5, c = i & 31;
        int m = mBase + r, k = k0 + c;
        bool ok = (m < M) && (k < K);
        As[r*40 + c] = (_Float16)(ok ? va[it] : 0.f);
      }
    }
    // ---- stage B (activations) 32(k) x 64(col), stored fragment-major
    if (MODE == 0) {
      float vb[16];
#pragma unroll
      for (int it = 0; it < 16; ++it) {
        int i = tid + it*128;
        int kr = i >> 6, c = i & 63;
        int k = k0 + kr, col = nBase + c;
        bool ok = (k < K) && (col < NC);
        long long a = (long long)b*xbs + (ok ? ((long long)k*ldx + col) : 0LL);
        vb[it] = X[a];
      }
#pragma unroll
      for (int it = 0; it < 16; ++it) {
        int i = tid + it*128;
        int kr = i >> 6, c = i & 63;
        int k = k0 + kr, col = nBase + c;
        bool ok = (k < K) && (col < NC);
        Bs[c*40 + kr] = (_Float16)(ok ? vb[it] : 0.f);
      }
    } else {
      float vb[16];
#pragma unroll
      for (int it = 0; it < 16; ++it) {
        int i = tid + it*128;
        int kr = i >> 6, c = i & 63;
        int k = k0 + kr;
        int col = nBase + c;
        int n = col >> 4;
        float v = 0.f;
        if (k < 960) {
          v = X[((long long)b*960 + k)*NPTS + n];                 // center feature
        } else if (k < 1920) {
          v = featT[((long long)b*960 + (k-960))*NPTS + nnCache[c]]; // neighbor
        } else if (k < K) {
          int d  = k - 1920;
          const float* xz = xyzT + ((long long)b*3 + d)*NPTS;
          v = xz[nnCache[c]] - xz[n];                             // relative xyz
        }
        vb[it] = v;
      }
#pragma unroll
      for (int it = 0; it < 16; ++it) {
        int i = tid + it*128;
        int kr = i >> 6, c = i & 63;
        Bs[c*40 + kr] = (_Float16)vb[it];
      }
    }
    __syncthreads();

    v16h a0 = frag_a(As, wm*32,      lane);
    v16h a1 = frag_a(As, wm*32 + 16, lane);
    v16h b0 = frag_b(Bs, wn*32,      lane);
    v16h b1 = frag_b(Bs, wn*32 + 16, lane);
    acc00 = __builtin_amdgcn_wmma_f32_16x16x32_f16(false, a0, false, b0, (short)0, acc00, false, false);
    acc01 = __builtin_amdgcn_wmma_f32_16x16x32_f16(false, a0, false, b1, (short)0, acc01, false, false);
    acc10 = __builtin_amdgcn_wmma_f32_16x16x32_f16(false, a1, false, b0, (short)0, acc10, false, false);
    acc11 = __builtin_amdgcn_wmma_f32_16x16x32_f16(false, a1, false, b1, (short)0, acc11, false, false);
  }

  const int nl = lane & 15, ro = (lane >> 4) * 8;
  auto st = [&](const v8f& a, int mt, int nt) {
#pragma unroll
    for (int r = 0; r < 8; ++r) {
      int m = mBase + mt + ro + r;
      int n = nBase + nt + nl;
      if (m < M && n < NC) {
        float v = a[r] + (bias ? bias[m] : 0.f);
        Y[(long long)b*ybs + (long long)m*ldy + n] = v;
      }
    }
  };
  st(acc00, wm*32,      wn*32);
  st(acc01, wm*32,      wn*32 + 16);
  st(acc10, wm*32 + 16, wn*32);
  st(acc11, wm*32 + 16, wn*32 + 16);
}

// ---------------------------------------------------------------------------
// BN: per-channel mean/var over (B x NC) -> scale/shift
// ---------------------------------------------------------------------------
__global__ void k_stats(const float* __restrict__ Y, long long ybs, int ldy, int NC, int B,
                        const float* __restrict__ g, const float* __restrict__ be,
                        float* __restrict__ scale, float* __restrict__ shift) {
  int c = blockIdx.x;
  __shared__ float s1[256], s2[256];
  float a = 0.f, q = 0.f;
  for (int b = 0; b < B; ++b) {
    const float* p = Y + (long long)b*ybs + (long long)c*ldy;
    for (int i = threadIdx.x; i < NC; i += 256) { float v = p[i]; a += v; q += v*v; }
  }
  s1[threadIdx.x] = a; s2[threadIdx.x] = q; __syncthreads();
  for (int s = 128; s > 0; s >>= 1) {
    if (threadIdx.x < s) { s1[threadIdx.x] += s1[threadIdx.x+s]; s2[threadIdx.x] += s2[threadIdx.x+s]; }
    __syncthreads();
  }
  if (threadIdx.x == 0) {
    float cnt  = (float)B * (float)NC;
    float mean = s1[0] / cnt;
    float var  = s2[0] / cnt - mean*mean;
    float sc   = g[c] * rsqrtf(var + 1e-5f);
    scale[c] = sc; shift[c] = be[c] - mean*sc;
  }
}

__global__ void k_affine(const float* __restrict__ src, float* __restrict__ dst,
                         long long bs, int ldy, int M, int NC,
                         const float* __restrict__ scale, const float* __restrict__ shift,
                         int relu, long long total) {
  long long i = (long long)blockIdx.x * blockDim.x + threadIdx.x;
  if (i >= total) return;
  int n = (int)(i % NC);
  long long t = i / NC;
  int r = (int)(t % M);
  int b = (int)(t / M);
  long long o = (long long)b*bs + (long long)r*ldy + n;
  float v = src[o]*scale[r] + shift[r];
  if (relu) v = fmaxf(v, 0.f);
  dst[o] = v;
}

// ---------------------------------------------------------------------------
// Small FC (ncols = B = 2): grid (O, B)
// ---------------------------------------------------------------------------
__global__ void k_fc(const float* __restrict__ W, const float* __restrict__ bias,
                     const float* __restrict__ X, int K, float* __restrict__ Y, int O) {
  int o = blockIdx.x, b = blockIdx.y;
  __shared__ float sh[128];
  const float* w = W + (long long)o*K;
  const float* x = X + (long long)b*K;
  float s = 0.f;
  for (int i = threadIdx.x; i < K; i += 128) s += w[i]*x[i];
  sh[threadIdx.x] = s; __syncthreads();
  for (int t = 64; t > 0; t >>= 1) { if (threadIdx.x < t) sh[threadIdx.x] += sh[threadIdx.x+t]; __syncthreads(); }
  if (threadIdx.x == 0) Y[(long long)b*O + o] = sh[0] + bias[o];
}

__global__ void k_bn_fc(float* __restrict__ Y, int O, int B,
                        const float* __restrict__ g, const float* __restrict__ be) {
  int c = blockIdx.x*256 + threadIdx.x;
  if (c >= O) return;
  float m = 0.f;
  for (int b = 0; b < B; ++b) m += Y[(long long)b*O + c];
  m /= (float)B;
  float v = 0.f;
  for (int b = 0; b < B; ++b) { float d = Y[(long long)b*O + c] - m; v += d*d; }
  v /= (float)B;
  float sc = g[c] * rsqrtf(v + 1e-5f);
  for (int b = 0; b < B; ++b) {
    float y = (Y[(long long)b*O + c] - m)*sc + be[c];
    Y[(long long)b*O + c] = fmaxf(y, 0.f);
  }
}

__global__ void k_maxpool(const float* __restrict__ H, int N, float* __restrict__ P, int O) {
  int o = blockIdx.x, b = blockIdx.y;
  __shared__ float sh[256];
  const float* p = H + ((long long)b*O + o)*N;
  float m = -1e30f;
  for (int i = threadIdx.x; i < N; i += 256) m = fmaxf(m, p[i]);
  sh[threadIdx.x] = m; __syncthreads();
  for (int t = 128; t > 0; t >>= 1) { if (threadIdx.x < t) sh[threadIdx.x] = fmaxf(sh[threadIdx.x], sh[threadIdx.x+t]); __syncthreads(); }
  if (threadIdx.x == 0) P[(long long)b*O + o] = sh[0];
}

__global__ void k_trans_finalize(const float* __restrict__ t9, float* __restrict__ transWS,
                                 float* __restrict__ outTrans) {
  int i = threadIdx.x;
  if (i < BATCH*9) {
    int e = i % 9;
    float v = t9[i] + ((e % 4 == 0) ? 1.f : 0.f);  // eye(3) at 0,4,8
    transWS[i] = v; outTrans[i] = v;
  }
}

// builds xyz_trans (+ output copy) and x_trans = [xyz_trans ; x[3:9]]
__global__ void k_build_xtrans(const float* __restrict__ x, const float* __restrict__ trans,
                               float* __restrict__ xt, float* __restrict__ xyzT,
                               float* __restrict__ outXyz) {
  int i = blockIdx.x*blockDim.x + threadIdx.x;
  if (i >= BATCH*NPTS) return;
  int b = i / NPTS, n = i % NPTS;
  const float* T = trans + b*9;
  const float* xb = x + (long long)b*9*NPTS;
  float p0 = xb[0*NPTS + n], p1 = xb[1*NPTS + n], p2 = xb[2*NPTS + n];
#pragma unroll
  for (int r = 0; r < 3; ++r) {
    float v = T[r*3+0]*p0 + T[r*3+1]*p1 + T[r*3+2]*p2;
    long long o = ((long long)b*3 + r)*NPTS + n;
    xyzT[o] = v; outXyz[o] = v;
    xt[((long long)b*9 + r)*NPTS + n] = v;
  }
#pragma unroll
  for (int r = 3; r < 9; ++r) xt[((long long)b*9 + r)*NPTS + n] = xb[r*NPTS + n];
}

// kNN: per point, 16 smallest squared distances (register bubble-insert, static idx)
__global__ void k_knn(const float* __restrict__ xyzT, int* __restrict__ idx) {
  int i = blockIdx.x*blockDim.x + threadIdx.x;
  if (i >= BATCH*NPTS) return;
  int b = i / NPTS, n = i % NPTS;
  const float* X = xyzT + (long long)b*3*NPTS;
  float px = X[n], py = X[NPTS + n], pz = X[2*NPTS + n];
  float bd[KNN]; int bi[KNN];
#pragma unroll
  for (int j = 0; j < KNN; ++j) { bd[j] = 1e30f; bi[j] = 0; }
  for (int m = 0; m < NPTS; ++m) {
    float dx = X[m]-px, dy = X[NPTS+m]-py, dz = X[2*NPTS+m]-pz;
    float d = dx*dx + dy*dy + dz*dz;
    if (d < bd[KNN-1]) {
      float cd = d; int ci = m;
#pragma unroll
      for (int j = 0; j < KNN; ++j) {
        if (cd < bd[j]) { float td = bd[j]; int ti = bi[j]; bd[j] = cd; bi[j] = ci; cd = td; ci = ti; }
      }
    }
  }
#pragma unroll
  for (int j = 0; j < KNN; ++j) idx[((long long)b*NPTS + n)*KNN + j] = bi[j];
}

// edge_logits = es_c2 over normalized es features: logits[b, n*16+kk]
__global__ void k_logits(const float* __restrict__ S1, const float* __restrict__ w,
                         const float* __restrict__ bias, float* __restrict__ logits) {
  long long i = (long long)blockIdx.x*blockDim.x + threadIdx.x;
  if (i >= (long long)BATCH*NKCOL) return;
  int b = (int)(i / NKCOL), j = (int)(i % NKCOL);
  const float* p = S1 + (long long)b*128*NKCOL + j;
  float s = bias[0];
  for (int c = 0; c < 128; ++c) s += w[c] * p[(long long)c*NKCOL];
  logits[i] = s;
}

// local_feat[b,c,n] = sum_k softmax_k(logits[b,n,:]) * lf[b,c,n,k]
__global__ void k_wsum(const float* __restrict__ E3, const float* __restrict__ logits,
                       float* __restrict__ local) {
  long long i = (long long)blockIdx.x*blockDim.x + threadIdx.x;
  if (i >= (long long)BATCH*64*NPTS) return;
  int n = (int)(i % NPTS);
  long long t = i / NPTS;
  int c = (int)(t % 64), b = (int)(t / 64);
  const float* L = logits + (long long)b*NKCOL + n*KNN;
  float l[KNN], mx = -1e30f;
#pragma unroll
  for (int k = 0; k < KNN; ++k) { l[k] = L[k]; mx = fmaxf(mx, l[k]); }
  float s = 0.f;
#pragma unroll
  for (int k = 0; k < KNN; ++k) { l[k] = __expf(l[k] - mx); s += l[k]; }
  float inv = 1.f / s;
  const float* F = E3 + ((long long)b*64 + c)*NKCOL + n*KNN;
  float acc = 0.f;
#pragma unroll
  for (int k = 0; k < KNN; ++k) acc += l[k]*inv * F[k];
  local[i] = acc;
}

__global__ void k_gate(float* __restrict__ local, const float* __restrict__ ng2, long long total) {
  long long i = (long long)blockIdx.x*blockDim.x + threadIdx.x;
  if (i >= total) return;
  local[i] *= 1.f / (1.f + __expf(-ng2[i]));
}

// affinity rows: s[n,m] = <p_n, p_m>; softmax over m
__global__ void k_aff_softmax(const float* __restrict__ xyzT, float* __restrict__ aff) {
  int n = blockIdx.x, b = blockIdx.y;
  __shared__ float sh[256];
  const float* X = xyzT + (long long)b*3*NPTS;
  float px = X[n], py = X[NPTS + n], pz = X[2*NPTS + n];
  float* row = aff + ((long long)b*NPTS + n)*NPTS;
  float mx = -1e30f;
  for (int m = threadIdx.x; m < NPTS; m += 256) {
    float s = px*X[m] + py*X[NPTS+m] + pz*X[2*NPTS+m];
    row[m] = s; mx = fmaxf(mx, s);
  }
  sh[threadIdx.x] = mx; __syncthreads();
  for (int t = 128; t > 0; t >>= 1) { if (threadIdx.x < t) sh[threadIdx.x] = fmaxf(sh[threadIdx.x], sh[threadIdx.x+t]); __syncthreads(); }
  mx = sh[0]; __syncthreads();
  float sum = 0.f;
  for (int m = threadIdx.x; m < NPTS; m += 256) { float e = __expf(row[m] - mx); row[m] = e; sum += e; }
  sh[threadIdx.x] = sum; __syncthreads();
  for (int t = 128; t > 0; t >>= 1) { if (threadIdx.x < t) sh[threadIdx.x] += sh[threadIdx.x+t]; __syncthreads(); }
  float inv = 1.f / sh[0]; __syncthreads();
  for (int m = threadIdx.x; m < NPTS; m += 256) row[m] *= inv;
}

// ---------------------------------------------------------------------------
extern "C" void kernel_launch(void* const* d_in, const int* in_sizes, int n_in,
                              void* d_out, int out_size, void* d_ws, size_t ws_size,
                              hipStream_t stream) {
  (void)in_sizes; (void)n_in; (void)out_size; (void)ws_size;
  const float* x = (const float*)d_in[0];
  const float *t_c1_w=(const float*)d_in[1],  *t_c1_b=(const float*)d_in[2];
  const float *t_bn1_g=(const float*)d_in[3], *t_bn1_b=(const float*)d_in[4];
  const float *t_c2_w=(const float*)d_in[5],  *t_c2_b=(const float*)d_in[6];
  const float *t_bn2_g=(const float*)d_in[7], *t_bn2_b=(const float*)d_in[8];
  const float *t_c3_w=(const float*)d_in[9],  *t_c3_b=(const float*)d_in[10];
  const float *t_bn3_g=(const float*)d_in[11],*t_bn3_b=(const float*)d_in[12];
  const float *t_f1_w=(const float*)d_in[13], *t_f1_b=(const float*)d_in[14];
  const float *t_bn4_g=(const float*)d_in[15],*t_bn4_b=(const float*)d_in[16];
  const float *t_f2_w=(const float*)d_in[17], *t_f2_b=(const float*)d_in[18];
  const float *t_bn5_g=(const float*)d_in[19],*t_bn5_b=(const float*)d_in[20];
  const float *t_f3_w=(const float*)d_in[21], *t_f3_b=(const float*)d_in[22];
  const float *c1_w=(const float*)d_in[23],   *c1_b=(const float*)d_in[24];
  const float *bn1_g=(const float*)d_in[25],  *bn1_b=(const float*)d_in[26];
  const float *c2_w=(const float*)d_in[27],   *c2_b=(const float*)d_in[28];
  const float *bn2_g=(const float*)d_in[29],  *bn2_b=(const float*)d_in[30];
  const float *c3_w=(const float*)d_in[31],   *c3_b=(const float*)d_in[32];
  const float *bn3_g=(const float*)d_in[33],  *bn3_b=(const float*)d_in[34];
  const float *c4_w=(const float*)d_in[35],   *c4_b=(const float*)d_in[36];
  const float *bn4_g=(const float*)d_in[37],  *bn4_b=(const float*)d_in[38];
  const float *e_c1_w=(const float*)d_in[39], *e_c1_b=(const float*)d_in[40];
  const float *e_bn1_g=(const float*)d_in[41],*e_bn1_b=(const float*)d_in[42];
  const float *e_c2_w=(const float*)d_in[43], *e_c2_b=(const float*)d_in[44];
  const float *e_bn2_g=(const float*)d_in[45],*e_bn2_b=(const float*)d_in[46];
  const float *e_c3_w=(const float*)d_in[47], *e_c3_b=(const float*)d_in[48];
  const float *e_bn3_g=(const float*)d_in[49],*e_bn3_b=(const float*)d_in[50];
  const float *es_c1_w=(const float*)d_in[51],*es_c1_b=(const float*)d_in[52];
  const float *es_bn_g=(const float*)d_in[53],*es_bn_b=(const float*)d_in[54];
  const float *es_c2_w=(const float*)d_in[55],*es_c2_b=(const float*)d_in[56];
  const float *ng_c1_w=(const float*)d_in[57],*ng_c1_b=(const float*)d_in[58];
  const float *ng_bn_g=(const float*)d_in[59],*ng_bn_b=(const float*)d_in[60];
  const float *ng_c2_w=(const float*)d_in[61],*ng_c2_b=(const float*)d_in[62];
  const float *bp_c_w=(const float*)d_in[63], *bp_c_b=(const float*)d_in[64];
  const float *bp_bn_g=(const float*)d_in[65],*bp_bn_b=(const float*)d_in[66];

  float* out = (float*)d_out;
  float* out_xyz   = out;                      // 2*3*2048
  float* out_topo  = out + 12288;              // 2*64*2048
  float* out_trans = out + 12288 + 262144;     // 18

  // workspace bump allocation (floats)
  float* Wf = (float*)d_ws;
  size_t off = 0;
  auto alloc = [&](size_t n){ float* p = Wf + off; off += n; return p; };
  const long long NN = NPTS, NK = NKCOL;
  float* h1    = alloc((size_t)BATCH*64*NN);
  float* h2    = alloc((size_t)BATCH*128*NN);
  float* h3    = alloc((size_t)BATCH*1024*NN);
  float* pool  = alloc((size_t)BATCH*1024);
  float* fcA   = alloc((size_t)BATCH*512);
  float* fcB   = alloc((size_t)BATCH*256);
  float* t9    = alloc((size_t)BATCH*9);
  float* trans = alloc((size_t)BATCH*9);
  float* xt    = alloc((size_t)BATCH*9*NN);
  float* xyzT  = alloc((size_t)BATCH*3*NN);
  float* featO = alloc((size_t)BATCH*960*NN);
  float* featT = alloc((size_t)BATCH*960*NN);
  int*   idxB  = (int*)alloc((size_t)BATCH*NN*KNN);
  float* E1    = alloc((size_t)BATCH*1024*NK);
  float* E2    = alloc((size_t)BATCH*128*NK);
  float* E3    = alloc((size_t)BATCH*64*NK);
  float* S1    = alloc((size_t)BATCH*128*NK);
  float* logit = alloc((size_t)BATCH*NK);
  float* localF= alloc((size_t)BATCH*64*NN);
  float* NG1   = alloc((size_t)BATCH*64*NN);
  float* NG2   = alloc((size_t)BATCH*64*NN);
  float* aff   = alloc((size_t)BATCH*NN*NN);
  float* TOPO  = alloc((size_t)BATCH*64*NN);
  float* BP    = alloc((size_t)BATCH*64*NN);
  float* scB   = alloc(1024);
  float* shB   = alloc(1024);

  auto gemmD = [&](const float* w, long long wbs, const float* bias,
                   const float* X, long long xbs, int ldx,
                   float* Y, long long ybs, int ldy, int M, int K, int NC) {
    dim3 g((NC+63)/64, (M+63)/64, BATCH);
    k_gemm<0><<<g, 128, 0, stream>>>(w, wbs, bias, X, xbs, ldx, Y, ybs, ldy, M, K, NC,
                                     nullptr, nullptr, nullptr);
  };
  auto gemmE = [&](const float* w, const float* bias, float* Y, int M) {
    dim3 g(NKCOL/64, (M+63)/64, BATCH);
    k_gemm<1><<<g, 128, 0, stream>>>(w, 0LL, bias, featO, 0LL, 0, Y, (long long)M*NK, NKCOL,
                                     M, EDGE_C, NKCOL, featT, idxB, xyzT);
  };
  auto bnAct = [&](float* Y, long long ybs, int ldy, int M, int NC,
                   const float* g, const float* be, int relu, float* dst) {
    k_stats<<<dim3(M), 256, 0, stream>>>(Y, ybs, ldy, NC, BATCH, g, be, scB, shB);
    long long tot = (long long)BATCH*M*NC;
    k_affine<<<dim3((unsigned)((tot+255)/256)), 256, 0, stream>>>(Y, dst, ybs, ldy, M, NC, scB, shB, relu, tot);
  };

  // ---------------- T-Net ----------------
  gemmD(t_c1_w, 0, t_c1_b, x, 9*NN, NPTS, h1, 64*NN, NPTS, 64, 3, NPTS);
  bnAct(h1, 64*NN, NPTS, 64, NPTS, t_bn1_g, t_bn1_b, 1, h1);
  gemmD(t_c2_w, 0, t_c2_b, h1, 64*NN, NPTS, h2, 128*NN, NPTS, 128, 64, NPTS);
  bnAct(h2, 128*NN, NPTS, 128, NPTS, t_bn2_g, t_bn2_b, 1, h2);
  gemmD(t_c3_w, 0, t_c3_b, h2, 128*NN, NPTS, h3, 1024*NN, NPTS, 1024, 128, NPTS);
  bnAct(h3, 1024*NN, NPTS, 1024, NPTS, t_bn3_g, t_bn3_b, 1, h3);
  k_maxpool<<<dim3(1024, BATCH), 256, 0, stream>>>(h3, NPTS, pool, 1024);
  k_fc<<<dim3(512, BATCH), 128, 0, stream>>>(t_f1_w, t_f1_b, pool, 1024, fcA, 512);
  k_bn_fc<<<dim3(2), 256, 0, stream>>>(fcA, 512, BATCH, t_bn4_g, t_bn4_b);
  k_fc<<<dim3(256, BATCH), 128, 0, stream>>>(t_f2_w, t_f2_b, fcA, 512, fcB, 256);
  k_bn_fc<<<dim3(1), 256, 0, stream>>>(fcB, 256, BATCH, t_bn5_g, t_bn5_b);
  k_fc<<<dim3(9, BATCH), 128, 0, stream>>>(t_f3_w, t_f3_b, fcB, 256, t9, 9);
  k_trans_finalize<<<dim3(1), 32, 0, stream>>>(t9, trans, out_trans);

  // transformed coords + x_trans
  k_build_xtrans<<<dim3((BATCH*NPTS+255)/256), 256, 0, stream>>>(x, trans, xt, xyzT, out_xyz);

  // ---------------- multilevel features ----------------
  auto multilevel = [&](const float* src, float* feat) {
    gemmD(c1_w, 0, c1_b, src, 9*NN, NPTS, feat + 0*NN, 960*NN, NPTS, 64, 9, NPTS);
    bnAct(feat + 0*NN, 960*NN, NPTS, 64, NPTS, bn1_g, bn1_b, 1, feat + 0*NN);
    gemmD(c2_w, 0, c2_b, feat + 0*NN, 960*NN, NPTS, feat + 64*NN, 960*NN, NPTS, 128, 64, NPTS);
    bnAct(feat + 64*NN, 960*NN, NPTS, 128, NPTS, bn2_g, bn2_b, 1, feat + 64*NN);
    gemmD(c3_w, 0, c3_b, feat + 64*NN, 960*NN, NPTS, feat + 192*NN, 960*NN, NPTS, 256, 128, NPTS);
    bnAct(feat + 192*NN, 960*NN, NPTS, 256, NPTS, bn3_g, bn3_b, 1, feat + 192*NN);
    gemmD(c4_w, 0, c4_b, feat + 192*NN, 960*NN, NPTS, feat + 448*NN, 960*NN, NPTS, 512, 256, NPTS);
    bnAct(feat + 448*NN, 960*NN, NPTS, 512, NPTS, bn4_g, bn4_b, 1, feat + 448*NN);
  };
  multilevel(x,  featO);
  multilevel(xt, featT);

  // ---------------- kNN + edge convs (fused gather into WMMA GEMM) ----------------
  k_knn<<<dim3((BATCH*NPTS+127)/128), 128, 0, stream>>>(xyzT, idxB);

  gemmE(e_c1_w, e_c1_b, E1, 1024);
  bnAct(E1, 1024*NK, NKCOL, 1024, NKCOL, e_bn1_g, e_bn1_b, 1, E1);
  gemmD(e_c2_w, 0, e_c2_b, E1, 1024*NK, NKCOL, E2, 128*NK, NKCOL, 128, 1024, NKCOL);
  bnAct(E2, 128*NK, NKCOL, 128, NKCOL, e_bn2_g, e_bn2_b, 1, E2);
  gemmD(e_c3_w, 0, e_c3_b, E2, 128*NK, NKCOL, E3, 64*NK, NKCOL, 64, 128, NKCOL);
  bnAct(E3, 64*NK, NKCOL, 64, NKCOL, e_bn3_g, e_bn3_b, 1, E3);

  gemmE(es_c1_w, es_c1_b, S1, 128);
  bnAct(S1, 128*NK, NKCOL, 128, NKCOL, es_bn_g, es_bn_b, 1, S1);
  k_logits<<<dim3((unsigned)(((long long)BATCH*NK+255)/256)), 256, 0, stream>>>(S1, es_c2_w, es_c2_b, logit);
  k_wsum<<<dim3((unsigned)(((long long)BATCH*64*NN+255)/256)), 256, 0, stream>>>(E3, logit, localF);

  // ---------------- neighborhood gate ----------------
  gemmD(ng_c1_w, 0, ng_c1_b, localF, 64*NN, NPTS, NG1, 64*NN, NPTS, 64, 64, NPTS);
  bnAct(NG1, 64*NN, NPTS, 64, NPTS, ng_bn_g, ng_bn_b, 1, NG1);
  gemmD(ng_c2_w, 0, ng_c2_b, NG1, 64*NN, NPTS, NG2, 64*NN, NPTS, 64, 64, NPTS);
  k_gate<<<dim3((unsigned)(((long long)BATCH*64*NN+255)/256)), 256, 0, stream>>>(localF, NG2, (long long)BATCH*64*NN);

  // ---------------- topology propagation ----------------
  k_aff_softmax<<<dim3(NPTS, BATCH), 256, 0, stream>>>(xyzT, aff);
  // topo[b,c,m] = sum_n localF[b,c,n] * aff[b,n,m]  (W = localF per batch)
  gemmD(localF, 64*NN, nullptr, aff, (long long)NN*NN, NPTS, TOPO, 64*NN, NPTS, 64, NPTS, NPTS);
  gemmD(bp_c_w, 0, bp_c_b, TOPO, 64*NN, NPTS, BP, 64*NN, NPTS, 64, 64, NPTS);
  bnAct(BP, 64*NN, NPTS, 64, NPTS, bp_bn_g, bp_bn_b, 1, out_topo);
}
